// PCgraph_19834158973336
// MI455X (gfx1250) — matrix operand order repaired and checked by reading, not compiled, not run on hardware
//
#include <hip/hip_runtime.h>
#include <hip/hip_bf16.h>

typedef __attribute__((ext_vector_type(16))) __bf16 v16bf;
typedef __attribute__((ext_vector_type(8)))  float  v8f;

constexpr int BATCH  = 128;
constexpr int NV     = 4096;
constexpr int TILE_N = 32;    // N columns per workgroup (2 wmma n-subtiles per wave)
constexpr int KOUTER = 128;   // K per outer iteration = 4 wmma k-steps
constexpr int LDS_PITCH = 272; // 256B of K data + 16B pad (68 dwords: conflict-free reads)

// ---------------------------------------------------------------------------
// ws layout (bytes):
//   A32   f32  [NV*NV]   @   0 MB  (64 MB)  dense edge-weight accumulation
//   Abf   bf16 [NV*NV]   @  64 MB  (32 MB)  A row-major
//   ATbf  bf16 [NV*NV]   @  96 MB  (32 MB)  A^T row-major
//   fxbf  bf16 [B*NV]    @ 128 MB  ( 1 MB)  tanh(x)
//   errbf bf16 [B*NV]    @ 129 MB  ( 1 MB)  error
//   dfdx  f32  [B*NV]    @ 130 MB  ( 2 MB)  1 - tanh(x)^2
// ---------------------------------------------------------------------------

__global__ __launch_bounds__(256) void edge_scatter_kernel(
    const int* __restrict__ ei, const float* __restrict__ w,
    float* __restrict__ A32, int E)
{
  int e = blockIdx.x * blockDim.x + threadIdx.x;
  if (e >= E) return;
  int s = ei[e];
  int t = ei[E + e];
  size_t idx = (size_t)s * NV + t;
  atomicAdd(A32 + idx, w[idx]);
}

__global__ __launch_bounds__(256) void prep_kernel(
    const float* __restrict__ x, const float* __restrict__ err,
    __bf16* __restrict__ fxbf, __bf16* __restrict__ errbf,
    float* __restrict__ dfdx)
{
  int i = blockIdx.x * blockDim.x + threadIdx.x;
  float fx = tanhf(x[i]);
  fxbf[i]  = (__bf16)fx;
  dfdx[i]  = 1.0f - fx * fx;
  errbf[i] = (__bf16)err[i];
}

__global__ __launch_bounds__(256) void convert_kernel(
    const float* __restrict__ A32, __bf16* __restrict__ Abf,
    __bf16* __restrict__ ATbf)
{
  int i = blockIdx.x * blockDim.x + threadIdx.x;
  int s = i >> 12;
  int t = i & (NV - 1);
  __bf16 b = (__bf16)A32[i];
  Abf[i] = b;
  ATbf[(size_t)t * NV + s] = b;   // transposed copy for the gradient GEMM
}

// One fused GEMM kernel; blockIdx.y selects:
//   y==0 : mu   = tanh(x) @ A^T          (Aop = fxbf,  Bmat = Abf)
//   y==1 : dEdx = err - dfdx*(err @ A)   (Aop = errbf, Bmat = ATbf)
// Wave w owns output rows [16w,16w+16); workgroup owns 32 output columns.
// B tile double-buffered in LDS (1 barrier / 8 WMMAs); fragment loads are
// software-pipelined one k-step deep so ds_load latency hides under WMMA.
__global__ __launch_bounds__(256) void pc_gemm_kernel(
    const __bf16* __restrict__ fxbf, const __bf16* __restrict__ errbf,
    const __bf16* __restrict__ Abf,  const __bf16* __restrict__ ATbf,
    const float* __restrict__ err32, const float* __restrict__ dfdx,
    float* __restrict__ out)
{
  __shared__ char ldsB[2 * TILE_N * LDS_PITCH];

  const int gemm = blockIdx.y;
  const __bf16* __restrict__ act  = gemm ? errbf : fxbf;
  const __bf16* __restrict__ Bmat = gemm ? ATbf  : Abf;

  const int tid    = threadIdx.x;
  const int wave   = tid >> 5;
  const int lane   = tid & 31;
  const int lane_m = lane & 15;
  const int khalf  = lane >> 4;
  const int m0     = wave * 16;
  const int n_block = blockIdx.x * TILE_N;

  v8f acc0 = {};
  v8f acc1 = {};

  // Cooperative fill: 256 threads x 2 x 16B = 32 rows x 256B (128 K of bf16)
  const int fr = tid >> 3;           // row within tile (0..31)
  const int fc = (tid & 7) * 16;     // byte offset of first chunk within row
  const __bf16* __restrict__ fillRow = Bmat + (size_t)(n_block + fr) * NV;

  char* buf0 = ldsB;
  char* buf1 = ldsB + TILE_N * LDS_PITCH;

  // prologue: stage first K-chunk into buf0
  {
    const char* src = (const char*)(fillRow + 0);
    *(uint4*)(buf0 + fr * LDS_PITCH + fc)       = *(const uint4*)(src + fc);
    *(uint4*)(buf0 + fr * LDS_PITCH + fc + 128) = *(const uint4*)(src + fc + 128);
  }
  __syncthreads();

  union Frag { uint4 u4[2]; v16bf v; };

  for (int k0 = 0; k0 < NV; k0 += KOUTER) {
    char* cur = ((k0 / KOUTER) & 1) ? buf1 : buf0;
    char* nxt = ((k0 / KOUTER) & 1) ? buf0 : buf1;

    // stage next K-chunk into the other buffer (uniform branch, all threads)
    if (k0 + KOUTER < NV) {
      const char* src = (const char*)(fillRow + (k0 + KOUTER));
      *(uint4*)(nxt + fr * LDS_PITCH + fc)       = *(const uint4*)(src + fc);
      *(uint4*)(nxt + fr * LDS_PITCH + fc + 128) = *(const uint4*)(src + fc + 128);
    }

    const __bf16* __restrict__ actRow = act + (size_t)(m0 + lane_m) * NV + k0;
    const char* bRow0 = cur + (lane_m)      * LDS_PITCH + khalf * 16;
    const char* bRow1 = cur + (16 + lane_m) * LDS_PITCH + khalf * 16;

    // two-deep fragment pipeline: load frags[kk+1] before wmma[kk]
    Frag af[2], b0f[2], b1f[2];

    // preload kk = 0
    af[0].u4[0]  = *(const uint4*)(actRow + khalf * 8);
    af[0].u4[1]  = *(const uint4*)(actRow + 16 + khalf * 8);
    b0f[0].u4[0] = *(const uint4*)(bRow0);
    b0f[0].u4[1] = *(const uint4*)(bRow0 + 32);
    b1f[0].u4[0] = *(const uint4*)(bRow1);
    b1f[0].u4[1] = *(const uint4*)(bRow1 + 32);

#pragma unroll
    for (int kk = 0; kk < 4; ++kk) {
      const int cb = kk & 1;
      const int nb = cb ^ 1;
      if (kk < 3) {  // issue next step's loads before this step's math
        const int kn = kk + 1;
        af[nb].u4[0]  = *(const uint4*)(actRow + kn * 32 + khalf * 8);
        af[nb].u4[1]  = *(const uint4*)(actRow + kn * 32 + 16 + khalf * 8);
        b0f[nb].u4[0] = *(const uint4*)(bRow0 + kn * 64);
        b0f[nb].u4[1] = *(const uint4*)(bRow0 + kn * 64 + 32);
        b1f[nb].u4[0] = *(const uint4*)(bRow1 + kn * 64);
        b1f[nb].u4[1] = *(const uint4*)(bRow1 + kn * 64 + 32);
      }
      acc0 = __builtin_amdgcn_wmma_f32_16x16x32_bf16(
          false, af[cb].v, false, b0f[cb].v, (short)0, acc0, false, false);
      acc1 = __builtin_amdgcn_wmma_f32_16x16x32_bf16(
          false, af[cb].v, false, b1f[cb].v, (short)0, acc1, false, false);
    }
    __syncthreads();   // single barrier per 8 WMMAs
  }

  // Epilogue. C/D layout: VGPR r, lane l -> M = r + 8*(l>>4), N = l&15.
#pragma unroll
  for (int r = 0; r < 8; ++r) {
    const int brow = m0 + r + 8 * khalf;
    {
      const int col = n_block + lane_m;
      const size_t idx = (size_t)brow * NV + col;
      if (gemm == 0) out[idx] = acc0[r];
      else out[(size_t)BATCH * NV + idx] = err32[idx] - dfdx[idx] * acc0[r];
    }
    {
      const int col = n_block + 16 + lane_m;
      const size_t idx = (size_t)brow * NV + col;
      if (gemm == 0) out[idx] = acc1[r];
      else out[(size_t)BATCH * NV + idx] = err32[idx] - dfdx[idx] * acc1[r];
    }
  }
}

extern "C" void kernel_launch(void* const* d_in, const int* in_sizes, int n_in,
                              void* d_out, int out_size, void* d_ws, size_t ws_size,
                              hipStream_t stream) {
  const float* x   = (const float*)d_in[0];
  const float* err = (const float*)d_in[1];
  const float* w   = (const float*)d_in[2];
  const int*   ei  = (const int*)d_in[3];
  const int E = in_sizes[3] / 2;

  char* ws = (char*)d_ws;
  const size_t MB = 1024ull * 1024ull;
  float*  A32   = (float*)(ws);
  __bf16* Abf   = (__bf16*)(ws + 64 * MB);
  __bf16* ATbf  = (__bf16*)(ws + 96 * MB);
  __bf16* fxbf  = (__bf16*)(ws + 128 * MB);
  __bf16* errbf = (__bf16*)(ws + 129 * MB);
  float*  dfdx  = (float*)(ws + 130 * MB);
  float*  out   = (float*)d_out;

  hipMemsetAsync(A32, 0, (size_t)NV * NV * sizeof(float), stream);
  edge_scatter_kernel<<<(E + 255) / 256, 256, 0, stream>>>(ei, w, A32, E);
  prep_kernel<<<(BATCH * NV) / 256, 256, 0, stream>>>(x, err, fxbf, errbf, dfdx);
  convert_kernel<<<(NV * NV) / 256, 256, 0, stream>>>(A32, Abf, ATbf);

  dim3 grid(NV / TILE_N, 2);
  pc_gemm_kernel<<<grid, 256, 0, stream>>>(fxbf, errbf, Abf, ATbf, err, dfdx, out);
}
